// TCERL_13254269075507
// MI455X (gfx1250) — matrix operand order repaired and checked by reading, not compiled, not run on hardware
//
#include <hip/hip_runtime.h>
#include <hip/hip_bf16.h>

typedef __attribute__((ext_vector_type(16))) _Float16 v16h;
typedef __attribute__((ext_vector_type(8)))  _Float16 v8h;
typedef __attribute__((ext_vector_type(8)))  float    v8f;

static inline unsigned divUp(unsigned a, unsigned b) { return (a + b - 1) / b; }

__device__ __forceinline__ float eluf(float x) { return x > 0.f ? x : expm1f(x); }

__device__ __forceinline__ v8f wmma16(v16h a, v16h b, v8f c) {
  return __builtin_amdgcn_wmma_f32_16x16x32_f16(false, a, false, b, (short)0, c, false, false);
}

// ---------------- f32 -> f16 convert ----------------
__global__ void k_f32_to_f16(const float* __restrict__ in, _Float16* __restrict__ out, int n) {
  int i = blockIdx.x * blockDim.x + threadIdx.x;
  if (i < n) out[i] = (_Float16)in[i];
}

// ---- pack conv weights w[co][ci][kw] (f32) directly into per-lane WMMA A fragments ----
// Contraction K = kw*64 + ci. Apack[((coTile*chunks + kc)*32 + lane)*16 + e], e = 2g+j,
// A layout: lane(half,m): elem 2g+j <-> M=coTile*16+m, K = kc*32 + half*16 + 2g + j
__global__ void k_pack_conv_a(const float* __restrict__ w, _Float16* __restrict__ ap,
                              int chunks, int KW) {
  int i = blockIdx.x * blockDim.x + threadIdx.x;
  int total = 4 * chunks * 32;
  if (i >= total) return;
  int lane = i & 31;
  int t = i >> 5;
  int kc = t % chunks;
  int coTile = t / chunks;
  int half = lane >> 4, m = lane & 15;
  int co = coTile * 16 + m;
#pragma unroll
  for (int e = 0; e < 16; ++e) {
    int g = e >> 1, j = e & 1;
    int K = kc * 32 + (half * 16 + 2 * g + j);
    int kw = K >> 6, ci = K & 63;
    ap[(size_t)i * 16 + e] = (_Float16)w[(co * 64 + ci) * KW + kw];
  }
}

// ---- pack GEMM weights W[K,N] (f32) into per-lane WMMA B fragments (zero-padded) ----
// B layout: lane(half,m): elem e <-> N=colTile*16+m, K = kc*32 + half*16 + e
__global__ void k_pack_gemm_b(const float* __restrict__ w, _Float16* __restrict__ bp,
                              int K, int N, int chunks, int colTiles) {
  int i = blockIdx.x * blockDim.x + threadIdx.x;
  int total = colTiles * chunks * 32;
  if (i >= total) return;
  int lane = i & 31;
  int t = i >> 5;
  int kc = t % chunks;
  int colTile = t / chunks;
  int half = lane >> 4, m = lane & 15;
  int col = colTile * 16 + m;
#pragma unroll
  for (int e = 0; e < 16; ++e) {
    int Kidx = kc * 32 + half * 16 + e;
    float v = (Kidx < K && col < N) ? w[(size_t)Kidx * N + col] : 0.f;
    bp[(size_t)i * 16 + e] = (_Float16)v;
  }
}

// ---------------- implicit-GEMM conv1d via WMMA, LDS-staged input ----------------
// Block = 128 threads = 4 waves = one (n, 64-wide l slab). Wave w: coPair = w&1, lPair = w>>1,
// computing a 2x2 grid of 16x16 tiles (2 co-tiles x 2 l-tiles): per K-chunk
// 2 A loads + 2 B LDS reads feed 4 WMMAs (2x arithmetic intensity vs 1 tile/wave).
__global__ void k_conv_wmma(const _Float16* __restrict__ inH,
                            const _Float16* __restrict__ Ap,
                            const float* __restrict__ bias,
                            const float* __restrict__ bng, const float* __restrict__ bnb,
                            const float* __restrict__ bnm, const float* __restrict__ bnv,
                            int useBN,
                            _Float16* __restrict__ outH, float* __restrict__ outF,
                            int Lin, int Lout, int lBlocks, int chunks, int stride, int pad) {
  __shared__ _Float16 ldsIn[148 * 64];   // [pos][ci], row = 128B (16B aligned)
  const int tid  = threadIdx.x;
  const int lane = tid & 31;
  const int half = lane >> 4;
  const int m    = lane & 15;
  const int cp   = (tid >> 5) & 1;       // co pair
  const int lp   = tid >> 6;             // l pair

  const int lb   = blockIdx.x % lBlocks;
  const int n    = blockIdx.x / lBlocks;
  const int l0   = lb * 64;
  const int KW   = chunks >> 1;
  const int NP   = stride * 63 + KW;     // positions needed by the 64-wide slab
  const int posBase = stride * l0 - pad;
  const _Float16* __restrict__ inN = inH + (size_t)n * 64 * Lin;

  // cooperative fill: thread i -> pos p = i>>3, ci block sub = i&7 (8 f16 = one b128 store)
  for (int i = tid; i < NP * 8; i += 128) {
    int p = i >> 3, sub = i & 7;
    int gpos = posBase + p;
    v8h v;
#pragma unroll
    for (int k = 0; k < 8; ++k) {
      int ci = sub * 8 + k;
      _Float16 x = (_Float16)0;
      if ((unsigned)gpos < (unsigned)Lin) x = inN[ci * Lin + gpos];
      v[k] = x;
    }
    ((v8h*)ldsIn)[i] = v;
  }
  __syncthreads();

  const _Float16* __restrict__ ap0 = Ap + (size_t)((2 * cp)     * chunks) * 32 * 16;
  const _Float16* __restrict__ ap1 = Ap + (size_t)((2 * cp + 1) * chunks) * 32 * 16;
  const int row0 = stride * ((2 * lp)     * 16 + m);
  const int row1 = stride * ((2 * lp + 1) * 16 + m);

  v8f acc00 = {}, acc01 = {}, acc10 = {}, acc11 = {};
#pragma unroll 2
  for (int kc = 0; kc < chunks; ++kc) {
    size_t aoff = ((size_t)kc * 32 + lane) * 16;
    v16h a0 = *(const v16h*)(ap0 + aoff);
    v16h a1 = *(const v16h*)(ap1 + aoff);
    int kw  = kc >> 1;
    int cio = ((kc & 1) << 5) + (half << 4);
    v16h b0 = *(const v16h*)&ldsIn[(row0 + kw) * 64 + cio];
    v16h b1 = *(const v16h*)&ldsIn[(row1 + kw) * 64 + cio];
    acc00 = wmma16(a0, b0, acc00);
    acc01 = wmma16(a0, b1, acc01);
    acc10 = wmma16(a1, b0, acc10);
    acc11 = wmma16(a1, b1, acc11);
  }

#pragma unroll
  for (int ct = 0; ct < 2; ++ct) {
#pragma unroll
    for (int lt = 0; lt < 2; ++lt) {
      const v8f& acc = ct == 0 ? (lt == 0 ? acc00 : acc01) : (lt == 0 ? acc10 : acc11);
      int l = l0 + (2 * lp + lt) * 16 + m;
      if (l >= Lout) continue;
#pragma unroll
      for (int r = 0; r < 8; ++r) {
        int co = (2 * cp + ct) * 16 + half * 8 + r;
        float v = acc[r] + bias[co];
        v = eluf(v);
        if (useBN) {
          float inv = bng[co] * rsqrtf(bnv[co] + 1e-5f);
          v = (v - bnm[co]) * inv + bnb[co];
        }
        size_t off = ((size_t)n * 64 + co) * Lout + l;
        if (outF) outF[off] = v;
        else      outH[off] = (_Float16)v;
      }
    }
  }
}

// ---- WMMA GEMM: C[M,N] = act(Ah[M,Kpad] x Wpack + bias); wave = rowTile x colPair ----
__global__ void k_gemm_wmma(const _Float16* __restrict__ Ah, const _Float16* __restrict__ Bp,
                            const float* __restrict__ bias,
                            _Float16* __restrict__ outH, float* __restrict__ outF,
                            int Kpad, int N, int colPairs, int act) {
  const int lane = threadIdx.x & 31;
  const int wave = threadIdx.x >> 5;
  const int half = lane >> 4;
  const int m    = lane & 15;
  const int tile = blockIdx.x * (blockDim.x >> 5) + wave;
  const int colPair = tile % colPairs;
  const int rowTile = tile / colPairs;
  const int chunks  = Kpad >> 5;

  const _Float16* __restrict__ arow = Ah + (size_t)(rowTile * 16 + m) * Kpad;
  const _Float16* __restrict__ bp0  = Bp + (size_t)((2 * colPair)     * chunks) * 32 * 16;
  const _Float16* __restrict__ bp1  = Bp + (size_t)((2 * colPair + 1) * chunks) * 32 * 16;

  v8f acc0 = {}, acc1 = {};
#pragma unroll 2
  for (int kc = 0; kc < chunks; ++kc) {
    v16h a;
#pragma unroll
    for (int g = 0; g < 8; ++g) {
      int kA = kc * 32 + (half * 16 + 2 * g);
      a[2 * g]     = arow[kA];
      a[2 * g + 1] = arow[kA + 1];
    }
    size_t boff = ((size_t)kc * 32 + lane) * 16;
    v16h b0 = *(const v16h*)(bp0 + boff);
    v16h b1 = *(const v16h*)(bp1 + boff);
    acc0 = wmma16(a, b0, acc0);
    acc1 = wmma16(a, b1, acc1);
  }

#pragma unroll
  for (int t = 0; t < 2; ++t) {
    const v8f& acc = t == 0 ? acc0 : acc1;
    int col = (2 * colPair + t) * 16 + m;
    if (col >= N) continue;
#pragma unroll
    for (int r = 0; r < 8; ++r) {
      int row = rowTile * 16 + half * 8 + r;
      float v = acc[r] + bias[col];
      if (act == 1) v = fmaxf(v, 0.f);
      else if (act == 2) v = eluf(v);
      size_t off = (size_t)row * N + col;
      if (outH) outH[off] = (_Float16)v;
      if (outF) outF[off] = v;
    }
  }
}

// ---------------- avg-pool window 3, stride 2 ----------------
__global__ void k_avgpool(const _Float16* __restrict__ in, _Float16* __restrict__ out,
                          int Lin, int Lout, int total) {
  int i = blockIdx.x * blockDim.x + threadIdx.x;
  if (i >= total) return;
  int l = i % Lout;
  int nc = i / Lout;
  const _Float16* p = in + (size_t)nc * Lin + 2 * l;
  float s = (float)p[0] + (float)p[1] + (float)p[2];
  out[i] = (_Float16)(s * (1.f / 3.f));
}

// ---------------- attention pooling over crops ----------------
__global__ void k_attention(const float* __restrict__ y4,
                            const float* __restrict__ attn_w,   // [35,256]
                            const float* __restrict__ attn_b,   // [256]
                            const float* __restrict__ attn_u,   // [256]
                            float* __restrict__ hpool) {        // [32,64,35]
  __shared__ float sh[16 * 35];
  __shared__ float contrib[16 * 256];
  __shared__ float alpha[16];
  int b = blockIdx.x >> 6;
  int c = blockIdx.x & 63;
  int tid = threadIdx.x;

  for (int i = tid; i < 16 * 35; i += 256) {
    int s = i / 35, t = i % 35;
    sh[i] = y4[(((size_t)(b * 16 + s)) * 64 + c) * 35 + t];
  }
  __syncthreads();
  {
    int d = tid;
    float bb = attn_b[d], uu = attn_u[d];
    for (int s = 0; s < 16; ++s) {
      float acc = bb;
      for (int t = 0; t < 35; ++t) acc += sh[s * 35 + t] * attn_w[t * 256 + d];
      contrib[s * 256 + d] = tanhf(acc) * uu;
    }
  }
  __syncthreads();
  if (tid < 16) {
    float sum = 0.f;
    for (int d = 0; d < 256; ++d) sum += contrib[tid * 256 + d];
    alpha[tid] = sum;
  }
  __syncthreads();
  if (tid == 0) {
    float mx = alpha[0];
    for (int s = 1; s < 16; ++s) mx = fmaxf(mx, alpha[s]);
    float den = 0.f;
    for (int s = 0; s < 16; ++s) { float e = expf(alpha[s] - mx); alpha[s] = e; den += e; }
    float inv = 1.f / den;
    for (int s = 0; s < 16; ++s) alpha[s] *= inv;
  }
  __syncthreads();
  if (tid < 35) {
    float o = 0.f;
    for (int s = 0; s < 16; ++s) o += sh[s * 35 + tid] * alpha[s];
    hpool[((size_t)b * 64 + c) * 35 + tid] = eluf(o);
  }
}

// ---- GIN aggregation: out[b,node,f] = h + sum over 64 nodes, f16 out padded to Fpad ----
__global__ void k_gin_agg(const float* __restrict__ inF, const _Float16* __restrict__ inH,
                          _Float16* __restrict__ out, int Fin, int Fpad) {
  int i = blockIdx.x * blockDim.x + threadIdx.x;
  if (i >= 32 * Fpad) return;
  int b = i / Fpad, f = i % Fpad;
  if (f >= Fin) {
    for (int node = 0; node < 64; ++node) out[((size_t)b * 64 + node) * Fpad + f] = (_Float16)0;
    return;
  }
  float vals[64];
  float s = 0.f;
  for (int node = 0; node < 64; ++node) {
    size_t off = ((size_t)b * 64 + node) * Fin + f;
    float v = inF ? inF[off] : (float)inH[off];
    vals[node] = v;
    s += v;
  }
  for (int node = 0; node < 64; ++node)
    out[((size_t)b * 64 + node) * Fpad + f] = (_Float16)(vals[node] + s);
}

// ---------------- small naive GEMMs (f32 A, and f16-A variant) ----------------
__global__ void k_gemm_f32(const float* __restrict__ A, const float* __restrict__ W,
                           const float* __restrict__ bias, float* __restrict__ C,
                           int M, int N, int K, int act) {
  int i = blockIdx.x * blockDim.x + threadIdx.x;
  if (i >= M * N) return;
  int row = i / N, col = i % N;
  float acc = bias[col];
  const float* a = A + (size_t)row * K;
  for (int k = 0; k < K; ++k) acc += a[k] * W[(size_t)k * N + col];
  if (act == 1) acc = fmaxf(acc, 0.f);
  else if (act == 2) acc = eluf(acc);
  C[i] = acc;
}

__global__ void k_gemm_h(const _Float16* __restrict__ A, const float* __restrict__ W,
                         const float* __restrict__ bias, float* __restrict__ C,
                         int M, int N, int K, int act) {
  int i = blockIdx.x * blockDim.x + threadIdx.x;
  if (i >= M * N) return;
  int row = i / N, col = i % N;
  float acc = bias[col];
  const _Float16* a = A + (size_t)row * K;
  for (int k = 0; k < K; ++k) acc += (float)a[k] * W[(size_t)k * N + col];
  if (act == 1) acc = fmaxf(acc, 0.f);
  else if (act == 2) acc = eluf(acc);
  C[i] = acc;
}

extern "C" void kernel_launch(void* const* d_in, const int* in_sizes, int n_in,
                              void* d_out, int out_size, void* d_ws, size_t ws_size,
                              hipStream_t stream) {
  const float* x    = (const float*)d_in[0];
  const float* w1   = (const float*)d_in[1];
  const float* b1   = (const float*)d_in[2];
  const float* bn1g = (const float*)d_in[3];
  const float* bn1b = (const float*)d_in[4];
  const float* bn1m = (const float*)d_in[5];
  const float* bn1v = (const float*)d_in[6];
  const float* w2   = (const float*)d_in[7];
  const float* b2   = (const float*)d_in[8];
  const float* bn2g = (const float*)d_in[9];
  const float* bn2b = (const float*)d_in[10];
  const float* bn2m = (const float*)d_in[11];
  const float* bn2v = (const float*)d_in[12];
  const float* w3   = (const float*)d_in[13];
  const float* b3   = (const float*)d_in[14];
  const float* w4   = (const float*)d_in[15];
  const float* b4   = (const float*)d_in[16];
  const float* atw  = (const float*)d_in[17];
  const float* atb  = (const float*)d_in[18];
  const float* atu  = (const float*)d_in[19];
  const float* g0w1 = (const float*)d_in[20];
  const float* g0b1 = (const float*)d_in[21];
  const float* g0w2 = (const float*)d_in[22];
  const float* g0b2 = (const float*)d_in[23];
  const float* g1w1 = (const float*)d_in[24];
  const float* g1b1 = (const float*)d_in[25];
  const float* g1w2 = (const float*)d_in[26];
  const float* g1b2 = (const float*)d_in[27];
  const float* g2w1 = (const float*)d_in[28];
  const float* g2b1 = (const float*)d_in[29];
  const float* g2w2 = (const float*)d_in[30];
  const float* g2b2 = (const float*)d_in[31];
  const float* cw1  = (const float*)d_in[32];
  const float* cb1  = (const float*)d_in[33];
  const float* cw2  = (const float*)d_in[34];
  const float* cb2  = (const float*)d_in[35];
  const float* cw3  = (const float*)d_in[36];
  const float* cb3  = (const float*)d_in[37];
  const float* cw4  = (const float*)d_in[38];
  const float* cb4  = (const float*)d_in[39];
  (void)in_sizes; (void)n_in; (void)out_size; (void)ws_size;

  char* ws = (char*)d_ws;
  size_t off = 0;
  auto carve = [&](size_t bytes) -> void* {
    void* p = ws + off;
    off += (bytes + 255) & ~(size_t)255;
    return p;
  };

  const int N = 512;  // 32 batches * 16 crops
  _Float16* xh    = (_Float16*)carve((size_t)N * 64 * 400 * 2);
  _Float16* ap1   = (_Float16*)carve((size_t)4 * 40 * 32 * 16 * 2);
  _Float16* ap2   = (_Float16*)carve((size_t)4 * 40 * 32 * 16 * 2);
  _Float16* ap3   = (_Float16*)carve((size_t)4 * 12 * 32 * 16 * 2);
  _Float16* ap4   = (_Float16*)carve((size_t)4 * 12 * 32 * 16 * 2);
  _Float16* y1    = (_Float16*)carve((size_t)N * 64 * 192 * 2);
  _Float16* y2    = (_Float16*)carve((size_t)N * 64 * 87 * 2);
  _Float16* y3    = (_Float16*)carve((size_t)N * 64 * 82 * 2);
  _Float16* y3p   = (_Float16*)carve((size_t)N * 64 * 40 * 2);
  float*    y4    = (float*)carve((size_t)N * 64 * 35 * 4);
  float*    hp    = (float*)carve((size_t)32 * 64 * 35 * 4);
  _Float16* bpG01 = (_Float16*)carve((size_t)16 * 2 * 32 * 16 * 2);   // g0_w1: K35->64, N256
  _Float16* bpG02 = (_Float16*)carve((size_t)16 * 8 * 32 * 16 * 2);   // g0_w2: K256, N256
  _Float16* bpG11 = (_Float16*)carve((size_t)16 * 8 * 32 * 16 * 2);
  _Float16* bpG12 = (_Float16*)carve((size_t)16 * 8 * 32 * 16 * 2);
  _Float16* bpG21 = (_Float16*)carve((size_t)4 * 8 * 32 * 16 * 2);    // g2_w1: K256, N35 (4 tiles, padded)
  _Float16* aggH  = (_Float16*)carve((size_t)2048 * 256 * 2);
  _Float16* tH    = (_Float16*)carve((size_t)2048 * 256 * 2);
  _Float16* hH    = (_Float16*)carve((size_t)2048 * 256 * 2);
  _Float16* t3H   = (_Float16*)carve((size_t)2048 * 35 * 2);
  float*    h3    = (float*)carve((size_t)2048 * 35 * 4);
  float*    fc1   = (float*)carve((size_t)32 * 256 * 4);
  float*    fc2   = (float*)carve((size_t)32 * 64 * 4);
  float*    fc3   = (float*)carve((size_t)32 * 16 * 4);

  // 1) convert input, pack conv-A and gemm-B fragments
  {
    int n = N * 64 * 400;
    k_f32_to_f16<<<divUp(n, 256), 256, 0, stream>>>(x, xh, n);
  }
  k_pack_conv_a<<<divUp(4 * 40 * 32, 256), 256, 0, stream>>>(w1, ap1, 40, 20);
  k_pack_conv_a<<<divUp(4 * 40 * 32, 256), 256, 0, stream>>>(w2, ap2, 40, 20);
  k_pack_conv_a<<<divUp(4 * 12 * 32, 256), 256, 0, stream>>>(w3, ap3, 12, 6);
  k_pack_conv_a<<<divUp(4 * 12 * 32, 256), 256, 0, stream>>>(w4, ap4, 12, 6);
  k_pack_gemm_b<<<divUp(16 * 2 * 32, 256), 256, 0, stream>>>(g0w1, bpG01, 35, 256, 2, 16);
  k_pack_gemm_b<<<divUp(16 * 8 * 32, 256), 256, 0, stream>>>(g0w2, bpG02, 256, 256, 8, 16);
  k_pack_gemm_b<<<divUp(16 * 8 * 32, 256), 256, 0, stream>>>(g1w1, bpG11, 256, 256, 8, 16);
  k_pack_gemm_b<<<divUp(16 * 8 * 32, 256), 256, 0, stream>>>(g1w2, bpG12, 256, 256, 8, 16);
  k_pack_gemm_b<<<divUp(4 * 8 * 32, 256), 256, 0, stream>>>(g2w1, bpG21, 256, 35, 8, 4);

  // 2) conv stack: block = (n, 64-wide l slab), 4 waves, each wave 2x2 tiles
  k_conv_wmma<<<N * 3, 128, 0, stream>>>(xh, ap1, b1, bn1g, bn1b, bn1m, bn1v, 1,
                                         y1, nullptr, 400, 192, 3, 40, 2, 1);
  k_conv_wmma<<<N * 2, 128, 0, stream>>>(y1, ap2, b2, bn2g, bn2b, bn2m, bn2v, 1,
                                         y2, nullptr, 192, 87, 2, 40, 2, 0);
  k_conv_wmma<<<N * 2, 128, 0, stream>>>(y2, ap3, b3, b3, b3, b3, b3, 0,
                                         y3, nullptr, 87, 82, 2, 12, 1, 0);
  {
    int total = N * 64 * 40;
    k_avgpool<<<divUp(total, 256), 256, 0, stream>>>(y3, y3p, 82, 40, total);
  }
  k_conv_wmma<<<N * 1, 128, 0, stream>>>(y3p, ap4, b4, b4, b4, b4, b4, 0,
                                         nullptr, y4, 40, 35, 1, 12, 1, 0);

  // 3) attention pooling -> hp [32,64,35]
  k_attention<<<32 * 64, 256, 0, stream>>>(y4, atw, atb, atu, hp);

  // 4) GIN via WMMA (f16 storage, f32 accumulate); wave = rowTile x colPair
  k_gin_agg<<<divUp(32 * 64, 256), 256, 0, stream>>>(hp, nullptr, aggH, 35, 64);
  k_gemm_wmma<<<(128 * 8) / 4, 128, 0, stream>>>(aggH, bpG01, g0b1, tH, nullptr, 64, 256, 8, 1);
  k_gemm_wmma<<<(128 * 8) / 4, 128, 0, stream>>>(tH, bpG02, g0b2, hH, nullptr, 256, 256, 8, 1);
  k_gin_agg<<<divUp(32 * 256, 256), 256, 0, stream>>>(nullptr, hH, aggH, 256, 256);
  k_gemm_wmma<<<(128 * 8) / 4, 128, 0, stream>>>(aggH, bpG11, g1b1, tH, nullptr, 256, 256, 8, 1);
  k_gemm_wmma<<<(128 * 8) / 4, 128, 0, stream>>>(tH, bpG12, g1b2, hH, nullptr, 256, 256, 8, 1);
  k_gin_agg<<<divUp(32 * 256, 256), 256, 0, stream>>>(nullptr, hH, aggH, 256, 256);
  k_gemm_wmma<<<(128 * 2) / 4, 128, 0, stream>>>(aggH, bpG21, g2b1, t3H, nullptr, 256, 35, 2, 1);
  k_gemm_h<<<divUp(2048 * 35, 256), 256, 0, stream>>>(t3H, g2w2, g2b2, h3, 2048, 35, 35, 0);

  // 5) classifier: h3 viewed as [32, 2240]
  k_gemm_f32<<<divUp(32 * 256, 256), 256, 0, stream>>>(h3, cw1, cb1, fc1, 32, 256, 2240, 2);
  k_gemm_f32<<<divUp(32 * 64, 256), 256, 0, stream>>>(fc1, cw2, cb2, fc2, 32, 64, 256, 2);
  k_gemm_f32<<<divUp(32 * 16, 256), 256, 0, stream>>>(fc2, cw3, cb3, fc3, 32, 16, 64, 2);
  k_gemm_f32<<<1, 64, 0, stream>>>(fc3, cw4, cb4, (float*)d_out, 32, 2, 16, 0);
}
